// Equivariant_Inter_62672162783759
// MI455X (gfx1250) — matrix operand order repaired and checked by reading, not compiled, not run on hardware
//
#include <hip/hip_runtime.h>
#include <math.h>

#define NN 25000
#define NE 400000
#define ND 128
#define ED 64
#define TD 128
#define HEADS 8

typedef __attribute__((ext_vector_type(16))) __bf16 v16bf;
typedef __attribute__((ext_vector_type(8)))  float  v8f;

__device__ __forceinline__ float siluf(float x){ return x / (1.0f + __expf(-x)); }

__device__ __forceinline__ v8f wmma_bf16(v16bf a, v16bf b, v8f c){
  // emits v_wmma_f32_16x16x32_bf16
  return __builtin_amdgcn_wmma_f32_16x16x32_bf16(false, a, false, b, (short)0, c, false, false);
}

// A-matrix fragment from a row-major bf16 LDS tile [16][stride].
// lanes 0-15: row m=lane, K = k0+{0..7} (elems 0-7), k0+{16..23} (elems 8-15);
// lanes 16-31: K = k0+{8..15}/{24..31}.
__device__ __forceinline__ v16bf load_a_lds(const __bf16* base, int stride, int k0, int lane){
  const __bf16* pr = base + (lane & 15) * stride + k0 + ((lane < 16) ? 0 : 8);
  v16bf a;
#pragma unroll
  for (int e = 0; e < 8; ++e){ a[e] = pr[e]; a[8 + e] = pr[16 + e]; }
  return a;
}

// B-matrix fragment from pre-packed weight buffer: blob per (nt,kc) is
// 32 lanes x 16 bf16 -> one coalesced 32B vector load per lane.
__device__ __forceinline__ v16bf load_b_frag(const __bf16* wf, int KC, int nt, int kc, int lane){
  return *(const v16bf*)(wf + ((((size_t)nt * KC + kc) * 32 + lane) << 4));
}

// D fragment (f32 16x16): VGPR r -> row r (lanes 0-15) / row r+8 (lanes 16-31), col = lane&15
__device__ __forceinline__ void store_d(float* tile, int stride, int c0, int lane, v8f d){
  int r0 = (lane < 16) ? 0 : 8;
  int c  = c0 + (lane & 15);
#pragma unroll
  for (int r = 0; r < 8; ++r) tile[(r0 + r) * stride + c] = d[r];
}

// [16,K](bf16 LDS) x [K,Nout](packed bf16 global) -> [16,Nout](f32 LDS).
// A fragments are hoisted into VGPRs (KC is compile-time) and reused across
// all N-tiles handled by this wave; 4 waves split the N-tiles.
template<int KC>
__device__ __forceinline__ void gemm_tile(const __bf16* aLds, int K,
                                          const __bf16* wf, int Nout,
                                          float* outLds, int tid){
  const int lane = tid & 31, wave = tid >> 5;
  const int NT = Nout >> 4;
  v16bf a[KC];
#pragma unroll
  for (int kc = 0; kc < KC; ++kc) a[kc] = load_a_lds(aLds, K, kc << 5, lane);
  for (int nt = wave; nt < NT; nt += 4){
    // prefetch first fragment of this wave's next N-tile (speculative, hi locality)
    __builtin_prefetch(wf + (((size_t)(nt + 4) * KC) * 32 + lane) * 16, 0, 3);
    v8f acc = {0.f,0.f,0.f,0.f,0.f,0.f,0.f,0.f};
#pragma unroll
    for (int kc = 0; kc < KC; ++kc){
      v16bf b = load_b_frag(wf, KC, nt, kc, lane);
      acc = wmma_bf16(a[kc], b, acc);
    }
    store_d(outLds, Nout, nt << 4, lane, acc);
  }
}

// ---- parallel per-row statistics: 8 threads per row, then 8-way combine ----
// part: LDS scratch of 16*8*2 floats. C must be a multiple of 8.
__device__ __forceinline__ void row_stats(const float* inT, int C, const float* bias,
                                          float eps, float* mu_s, float* rs_s,
                                          float* part, int tid)
{
  int r = tid >> 3, t = tid & 7;
  int chunk = C >> 3;
  const float* row = inT + r * C;
  float s = 0.f, s2 = 0.f;
  for (int c = t * chunk; c < (t + 1) * chunk; ++c){
    float v = row[c] + (bias ? bias[c] : 0.f);
    s += v; s2 += v * v;
  }
  part[(r * 8 + t) * 2]     = s;
  part[(r * 8 + t) * 2 + 1] = s2;
  __syncthreads();
  if (tid < 16){
    float S = 0.f, S2 = 0.f;
#pragma unroll
    for (int t2 = 0; t2 < 8; ++t2){ S += part[(tid*8 + t2)*2]; S2 += part[(tid*8 + t2)*2 + 1]; }
    float inv = 1.f / (float)C;
    float mu = S * inv, var = S2 * inv - mu * mu;
    mu_s[tid] = mu; rs_s[tid] = rsqrtf(var + eps);
  }
  __syncthreads();
}

// raw per-row sum / sumsq (no eps), same parallel scheme
__device__ __forceinline__ void row_sums(const float* inT, int C,
                                         float* S_out, float* S2_out,
                                         float* part, int tid)
{
  int r = tid >> 3, t = tid & 7;
  int chunk = C >> 3;
  const float* row = inT + r * C;
  float s = 0.f, s2 = 0.f;
  for (int c = t * chunk; c < (t + 1) * chunk; ++c){
    float v = row[c]; s += v; s2 += v * v;
  }
  part[(r * 8 + t) * 2]     = s;
  part[(r * 8 + t) * 2 + 1] = s2;
  __syncthreads();
  if (tid < 16){
    float S = 0.f, S2 = 0.f;
#pragma unroll
    for (int t2 = 0; t2 < 8; ++t2){ S += part[(tid*8 + t2)*2]; S2 += part[(tid*8 + t2)*2 + 1]; }
    S_out[tid] = S; S2_out[tid] = S2;
  }
  __syncthreads();
}

// per-row LayerNorm (+bias, gamma/beta, SiLU) of f32 LDS tile [16][C] -> bf16 tile
__device__ __forceinline__ void ln_act_rows(const float* inT, int C,
    const float* bias, const float* g, const float* b, float eps, int do_silu,
    __bf16* outT, float* mu_s, float* rs_s, float* part, int tid)
{
  row_stats(inT, C, bias, eps, mu_s, rs_s, part, tid);
  for (int i = tid; i < 16 * C; i += 128){
    int r = i / C, c = i - r * C;
    float v = inT[i] + bias[c];
    v = (v - mu_s[r]) * rs_s[r];
    v = v * g[c] + b[c];
    if (do_silu) v = v / (1.f + __expf(-v));
    outT[i] = (__bf16)v;
  }
  __syncthreads();
}

// ordered-uint encoding for float atomicMax
__device__ __forceinline__ unsigned fenc(float f){
  unsigned u = __float_as_uint(f);
  return (u & 0x80000000u) ? ~u : (u | 0x80000000u);
}
__device__ __forceinline__ float fdec(unsigned u){
  u = (u & 0x80000000u) ? (u & 0x7fffffffu) : ~u;
  return __uint_as_float(u);
}

// ---------------- weight repack: (k,n) fp32 -> WMMA B-frag bf16 ----------------
__global__ void k_wconv(const float* __restrict__ w, __bf16* __restrict__ wf, int K, int Nout){
  int idx = blockIdx.x * 256 + threadIdx.x;
  if (idx >= K * Nout) return;
  int k = idx / Nout, n = idx - k * Nout;
  int nt = n >> 4, nn = n & 15, kc = k >> 5, kk = k & 31;
  int lane = nn + ((kk & 8) ? 16 : 0);
  int e = (kk & 7) + ((kk & 16) ? 8 : 0);
  int KC = K >> 5;
  wf[((((size_t)nt * KC + kc) * 32 + lane) << 4) + e] = (__bf16)w[idx];
}

// ---------------- init accumulators ----------------
__global__ void k_init(float* outnum, float* segsum, unsigned* segmax){
  int idx = blockIdx.x * 256 + threadIdx.x;
  if (idx < NN * ND) outnum[idx] = 0.f;
  if (idx < NN * HEADS){ segsum[idx] = 0.f; segmax[idx] = 0u; }
}

// ---------------- node pass 1: adaLN + hm + sums + q-MLP ----------------
__global__ __launch_bounds__(128)
void k_node_a(const float* __restrict__ h, const float* __restrict__ ntime,
              const float* __restrict__ nt_b,
              const float* __restrict__ q1_b, const float* __restrict__ qn_g,
              const float* __restrict__ qn_b, const float* __restrict__ q2_b,
              const __bf16* __restrict__ wf_nt, const __bf16* __restrict__ wf_q1,
              const __bf16* __restrict__ wf_q2,
              __bf16* __restrict__ hmbf, float* __restrict__ nodesums, float* __restrict__ qbuf)
{
  __shared__ __attribute__((aligned(16))) __bf16 sA[16*128];
  __shared__ __attribute__((aligned(16))) float  sNtm[16*256];
  __shared__ __attribute__((aligned(16))) float  sH[16*128];
  __shared__ __attribute__((aligned(16))) __bf16 sHm[16*128];
  __shared__ __attribute__((aligned(16))) float  sF[16*256];
  __shared__ __attribute__((aligned(16))) __bf16 sB[16*256];
  __shared__ float sMu[16], sRs[16];
  __shared__ float sP[16*8*2];
  const int tid = threadIdx.x;
  const int n0 = blockIdx.x * 16;

  for (int i = tid; i < 16*128; i += 128){
    int r = i >> 7, c = i & 127;
    int node = n0 + r; int rr = node < NN ? node : NN - 1;
    sA[i] = (__bf16)siluf(ntime[(size_t)rr * TD + c]);
    sH[i] = h[(size_t)rr * ND + c];
  }
  __syncthreads();
  gemm_tile<4>(sA, 128, wf_nt, 256, sNtm, tid);      // node_tm (pre-bias)
  __syncthreads();
  row_stats(sH, 128, nullptr, 1e-6f, sMu, sRs, sP, tid);   // LN(h) eps 1e-6
  for (int i = tid; i < 16*128; i += 128){           // hm = modulate(ln(h))
    int r = i >> 7, c = i & 127;
    float shift = sNtm[r*256 + c] + nt_b[c];
    float scale = sNtm[r*256 + 128 + c] + nt_b[128 + c];
    float lnh = (sH[i] - sMu[r]) * sRs[r];
    __bf16 hb = (__bf16)(lnh * (1.f + scale) + shift);
    sHm[i] = hb;
    sH[i] = (float)hb;                                // rounded value for sums
    int node = n0 + r;
    if (node < NN) hmbf[(size_t)node * ND + c] = hb;
  }
  __syncthreads();
  row_sums(sH, 128, sMu, sRs, sP, tid);              // per-node sums for x_feat LN
  if (tid < 16){
    int node = n0 + tid;
    if (node < NN){ nodesums[node*2] = sMu[tid]; nodesums[node*2 + 1] = sRs[tid]; }
  }
  // q = silu(LN(hm@q1+b)) @ q2 + b
  gemm_tile<4>(sHm, 128, wf_q1, 256, sF, tid);
  __syncthreads();
  ln_act_rows(sF, 256, q1_b, qn_g, qn_b, 1e-5f, 1, sB, sMu, sRs, sP, tid);
  gemm_tile<8>(sB, 256, wf_q2, 128, sF, tid);
  __syncthreads();
  for (int i = tid; i < 16*128; i += 128){
    int r = i >> 7, c = i & 127;
    int node = n0 + r;
    if (node < NN) qbuf[(size_t)node * ND + c] = sF[r*128 + c] + q2_b[c];
  }
}

// ---------------- edge pass 1: emb, x_feat, k/v MLPs, logits ----------------
__global__ __launch_bounds__(128)
void k_edge_b(const long long* __restrict__ eidx, const float* __restrict__ pos,
              const float* __restrict__ clash, const float* __restrict__ eattr,
              const float* __restrict__ etime,
              const float* __restrict__ et_b, const float* __restrict__ ee_b,
              const float* __restrict__ ln_g, const float* __restrict__ ln_b,
              const float* __restrict__ k1_b, const float* __restrict__ kn_g,
              const float* __restrict__ kn_b, const float* __restrict__ k2_b,
              const float* __restrict__ v1_b, const float* __restrict__ vn_g,
              const float* __restrict__ vn_b, const float* __restrict__ v2_b,
              const __bf16* __restrict__ wf_et, const __bf16* __restrict__ wf_ee,
              const __bf16* __restrict__ wf_k1, const __bf16* __restrict__ wf_k2,
              const __bf16* __restrict__ wf_v1, const __bf16* __restrict__ wf_v2,
              const __bf16* __restrict__ hmbf, const float* __restrict__ nodesums,
              const float* __restrict__ qbuf,
              __bf16* __restrict__ vbuf, float* __restrict__ logits, unsigned* __restrict__ segmax)
{
  __shared__ int sSrc[16], sDst[16];
  __shared__ float sDist[16];
  __shared__ __attribute__((aligned(16))) __bf16 sCat[16*192];
  __shared__ __attribute__((aligned(16))) __bf16 sX[16*320];
  __shared__ __attribute__((aligned(16))) float  sF[16*256];
  __shared__ __attribute__((aligned(16))) __bf16 sB[16*256];
  __shared__ __attribute__((aligned(16))) float  sEtm[16*128];
  __shared__ __attribute__((aligned(16))) float  sEm[16*64];
  __shared__ __attribute__((aligned(16))) float  sK[16*128];
  __shared__ float sMu[16], sRs[16];
  __shared__ float sP[16*8*2];

  const int tid = threadIdx.x;
  const long long e0 = (long long)blockIdx.x * 16;

  if (tid < 16){
    long long e = e0 + tid;
    int si = (int)eidx[e];
    int di = (int)eidx[NE + e];
    sSrc[tid] = si; sDst[tid] = di;
    float dx = pos[si*3+0] - pos[di*3+0];
    float dy = pos[si*3+1] - pos[di*3+1];
    float dz = pos[si*3+2] - pos[di*3+2];
    sDist[tid] = sqrtf(dx*dx + dy*dy + dz*dz);
  }
  for (int i = tid; i < 16*128; i += 128){           // silu(edge_time_emb)
    int r = i >> 7, c = i & 127;
    sB[i] = (__bf16)siluf(etime[(e0 + r) * TD + c]);
  }
  __syncthreads();
  gemm_tile<4>(sB, 128, wf_et, 128, sEtm, tid);      // edge_tm (pre-bias)

  const float gstep = 15.f / 63.f;
  const float gcoef = -0.5f / (gstep * gstep);
  for (int i = tid; i < 16*192; i += 128){           // [clash | gauss(dist) | edge_attr]
    int r = i / 192, c = i - r * 192;
    float v;
    if (c < 64)        v = clash[(e0 + r) * ED + c];
    else if (c < 128){ float dd = sDist[r] - (float)(c - 64) * gstep; v = __expf(gcoef * dd * dd); }
    else               v = eattr[(e0 + r) * ED + (c - 128)];
    sCat[i] = (__bf16)v;
  }
  __syncthreads();
  gemm_tile<6>(sCat, 192, wf_ee, 64, sF, tid);       // edge_emb (pre-bias)
  __syncthreads();
  row_stats(sF, 64, ee_b, 1e-6f, sMu, sRs, sP, tid); // LN(edge_emb) eps 1e-6
  for (int i = tid; i < 16*64; i += 128){            // em = modulate(ln(edge_emb))
    int r = i >> 6, c = i & 63;
    float lnv = (sF[r*64 + c] + ee_b[c] - sMu[r]) * sRs[r];
    float shift = sEtm[r*128 + c] + et_b[c];
    float scale = sEtm[r*128 + 64 + c] + et_b[64 + c];
    sEm[i] = lnv * (1.f + scale) + shift;
  }
  __syncthreads();
  row_sums(sEm, 64, sMu, sRs, sP, tid);              // raw em sums
  if (tid < 16){                                     // x_feat LN stats over 320
    int s0 = sSrc[tid], d0 = sDst[tid];
    float S  = sMu[tid] + nodesums[s0*2]     + nodesums[d0*2];
    float S2 = sRs[tid] + nodesums[s0*2 + 1] + nodesums[d0*2 + 1];
    float m = S * (1.f/320.f);
    float v = S2 * (1.f/320.f) - m * m;
    sMu[tid] = m; sRs[tid] = rsqrtf(v + 1e-5f);
  }
  __syncthreads();
  for (int i = tid; i < 16*320; i += 128){           // x_feat (bf16), gathers hm[src]/hm[dst]
    int r = i / 320, c = i - r * 320;
    float raw;
    if (c < 64)       raw = sEm[r*64 + c];
    else if (c < 192) raw = (float)hmbf[(size_t)sSrc[r] * ND + (c - 64)];
    else              raw = (float)hmbf[(size_t)sDst[r] * ND + (c - 192)];
    float v = (raw - sMu[r]) * sRs[r] * ln_g[c] + ln_b[c];
    sX[i] = (__bf16)v;
  }
  __syncthreads();
  // k = silu(LN(x@k1+b)) @ k2 + b
  gemm_tile<10>(sX, 320, wf_k1, 128, sF, tid);
  __syncthreads();
  ln_act_rows(sF, 128, k1_b, kn_g, kn_b, 1e-5f, 1, sB, sMu, sRs, sP, tid);
  gemm_tile<4>(sB, 128, wf_k2, 128, sK, tid);
  __syncthreads();
  {                                                  // logits + seg max (16 edges x 8 heads)
    int r = tid >> 3, hh = tid & 7;
    const float* qrow = qbuf + (size_t)sDst[r] * ND + hh * 16;
    float acc = 0.f;
#pragma unroll
    for (int j = 0; j < 16; ++j) acc += qrow[j] * (sK[r*128 + hh*16 + j] + k2_b[hh*16 + j]);
    acc *= 0.25f;                                    // 1/sqrt(HEAD_DIM)
    logits[(e0 + r) * HEADS + hh] = acc;
    atomicMax(&segmax[sDst[r] * HEADS + hh], fenc(acc));
  }
  __syncthreads();
  // v = silu(LN(x@v1+b)) @ v2 + b
  gemm_tile<10>(sX, 320, wf_v1, 256, sF, tid);
  __syncthreads();
  ln_act_rows(sF, 256, v1_b, vn_g, vn_b, 1e-5f, 1, sB, sMu, sRs, sP, tid);
  gemm_tile<8>(sB, 256, wf_v2, 128, sK, tid);
  __syncthreads();
  for (int i = tid; i < 16*128; i += 128){
    int r = i >> 7, c = i & 127;
    vbuf[(e0 + r) * ND + c] = (__bf16)(sK[i] + v2_b[c]);
  }
}

// ---------------- edge pass 2: exp + scatter-add numerator/denominator ----------------
__global__ __launch_bounds__(256)
void k_edge_c(const long long* __restrict__ eidx, const float* __restrict__ logits,
              const unsigned* __restrict__ segmax, const __bf16* __restrict__ vbuf,
              float* __restrict__ segsum, float* __restrict__ outnum)
{
  long long idx = (long long)blockIdx.x * 256 + threadIdx.x;
  if (idx >= (long long)NE * HEADS) return;
  long long e = idx >> 3; int hh = (int)(idx & 7);
  int d = (int)eidx[NE + e];
  float mx = fdec(segmax[d * HEADS + hh]);
  float ex = __expf(logits[e * HEADS + hh] - mx);
  atomicAdd(&segsum[d * HEADS + hh], ex);
  const __bf16* vp = vbuf + e * ND + hh * 16;
  float* op = outnum + (size_t)d * ND + hh * 16;
#pragma unroll
  for (int j = 0; j < 16; ++j) atomicAdd(&op[j], ex * (float)vp[j]);
}

// ---------------- node pass 2: softmax divide, o-MLP, FFN, residual ----------------
__global__ __launch_bounds__(128)
void k_node_d(const float* __restrict__ h, const float* __restrict__ outnum,
              const float* __restrict__ segsum,
              const float* __restrict__ o1_b, const float* __restrict__ o2_b,
              const float* __restrict__ ff1_b, const float* __restrict__ ff2_b,
              const __bf16* __restrict__ wf_o1, const __bf16* __restrict__ wf_o2,
              const __bf16* __restrict__ wf_ff1, const __bf16* __restrict__ wf_ff2,
              float* __restrict__ out)
{
  __shared__ __attribute__((aligned(16))) __bf16 sA[16*128];
  __shared__ __attribute__((aligned(16))) __bf16 sB[16*256];
  __shared__ __attribute__((aligned(16))) float  sF[16*256];
  const int tid = threadIdx.x;
  const int n0 = blockIdx.x * 16;

  for (int i = tid; i < 16*128; i += 128){
    int r = i >> 7, c = i & 127;
    int node = n0 + r; int rr = node < NN ? node : NN - 1;
    float v = outnum[(size_t)rr * ND + c] / (segsum[rr * HEADS + (c >> 4)] + 1e-16f);
    sA[i] = (__bf16)v;
  }
  __syncthreads();
  gemm_tile<4>(sA, 128, wf_o1, 256, sF, tid);
  __syncthreads();
  for (int i = tid; i < 16*256; i += 128){ int c = i & 255; sB[i] = (__bf16)siluf(sF[i] + o1_b[c]); }
  __syncthreads();
  gemm_tile<8>(sB, 256, wf_o2, 128, sF, tid);
  __syncthreads();
  for (int i = tid; i < 16*128; i += 128){ int c = i & 127; sA[i] = (__bf16)(sF[i] + o2_b[c]); }
  __syncthreads();
  gemm_tile<4>(sA, 128, wf_ff1, 256, sF, tid);
  __syncthreads();
  for (int i = tid; i < 16*256; i += 128){ int c = i & 255; sB[i] = (__bf16)siluf(sF[i] + ff1_b[c]); }
  __syncthreads();
  gemm_tile<8>(sB, 256, wf_ff2, 128, sF, tid);
  __syncthreads();
  for (int i = tid; i < 16*128; i += 128){
    int r = i >> 7, c = i & 127;
    int node = n0 + r;
    if (node < NN)
      out[(size_t)node * ND + c] = h[(size_t)node * ND + c] + sF[i] + ff2_b[c];
  }
}

__global__ void k_pos(const float* __restrict__ pos, float* __restrict__ out){
  int idx = blockIdx.x * 256 + threadIdx.x;
  if (idx < NN * 3) out[NN * ND + idx] = pos[idx];
}

extern "C" void kernel_launch(void* const* d_in, const int* in_sizes, int n_in,
                              void* d_out, int out_size, void* d_ws, size_t ws_size,
                              hipStream_t stream)
{
  (void)in_sizes; (void)n_in; (void)out_size; (void)ws_size;
  const float* pos   = (const float*)d_in[0];
  const float* h     = (const float*)d_in[1];
  const float* eattr = (const float*)d_in[2];
  const float* clash = (const float*)d_in[3];
  const long long* eidx = (const long long*)d_in[4];
  const float* ntime = (const float*)d_in[5];
  const float* etime = (const float*)d_in[6];
  const float* ee_w = (const float*)d_in[7];  const float* ee_b = (const float*)d_in[8];
  const float* ln_g = (const float*)d_in[9];  const float* ln_b = (const float*)d_in[10];
  const float* q1w = (const float*)d_in[11];  const float* q1b = (const float*)d_in[12];
  const float* qng = (const float*)d_in[13];  const float* qnb = (const float*)d_in[14];
  const float* q2w = (const float*)d_in[15];  const float* q2b = (const float*)d_in[16];
  const float* k1w = (const float*)d_in[17];  const float* k1b = (const float*)d_in[18];
  const float* kng = (const float*)d_in[19];  const float* knb = (const float*)d_in[20];
  const float* k2w = (const float*)d_in[21];  const float* k2b = (const float*)d_in[22];
  const float* v1w = (const float*)d_in[23];  const float* v1b = (const float*)d_in[24];
  const float* vng = (const float*)d_in[25];  const float* vnb = (const float*)d_in[26];
  const float* v2w = (const float*)d_in[27];  const float* v2b = (const float*)d_in[28];
  const float* o1w = (const float*)d_in[29];  const float* o1b = (const float*)d_in[30];
  const float* o2w = (const float*)d_in[31];  const float* o2b = (const float*)d_in[32];
  const float* ntw = (const float*)d_in[33];  const float* ntb = (const float*)d_in[34];
  const float* etw = (const float*)d_in[35];  const float* etb = (const float*)d_in[36];
  const float* f1w = (const float*)d_in[37];  const float* f1b = (const float*)d_in[38];
  const float* f2w = (const float*)d_in[39];  const float* f2b = (const float*)d_in[40];
  float* out = (float*)d_out;

  char* ws = (char*)d_ws;
  size_t off = 0;
  auto take = [&](size_t bytes)->char*{
    char* p = ws + off; off = (off + bytes + 255) & ~(size_t)255; return p;
  };
  __bf16* wfee = (__bf16*)take((size_t)192*64*2);
  __bf16* wfk1 = (__bf16*)take((size_t)320*128*2);
  __bf16* wfk2 = (__bf16*)take((size_t)128*128*2);
  __bf16* wfv1 = (__bf16*)take((size_t)320*256*2);
  __bf16* wfv2 = (__bf16*)take((size_t)256*128*2);
  __bf16* wfq1 = (__bf16*)take((size_t)128*256*2);
  __bf16* wfq2 = (__bf16*)take((size_t)256*128*2);
  __bf16* wfo1 = (__bf16*)take((size_t)128*256*2);
  __bf16* wfo2 = (__bf16*)take((size_t)256*128*2);
  __bf16* wfnt = (__bf16*)take((size_t)128*256*2);
  __bf16* wfet = (__bf16*)take((size_t)128*128*2);
  __bf16* wff1 = (__bf16*)take((size_t)128*256*2);
  __bf16* wff2 = (__bf16*)take((size_t)256*128*2);
  __bf16* hmbf     = (__bf16*)take((size_t)NN*ND*2);
  float*  nodesums = (float*) take((size_t)NN*2*4);
  float*  qbuf     = (float*) take((size_t)NN*ND*4);
  __bf16* vbuf     = (__bf16*)take((size_t)NE*ND*2);
  float*  logits   = (float*) take((size_t)NE*HEADS*4);
  unsigned* segmax = (unsigned*)take((size_t)NN*HEADS*4);
  float*  segsum   = (float*) take((size_t)NN*HEADS*4);
  float*  outnum   = (float*) take((size_t)NN*ND*4);

#define WCONV(S, D, K, Nn) k_wconv<<<((K)*(Nn)+255)/256, 256, 0, stream>>>(S, D, K, Nn)
  WCONV(ee_w, wfee, 192, 64);
  WCONV(k1w,  wfk1, 320, 128);
  WCONV(k2w,  wfk2, 128, 128);
  WCONV(v1w,  wfv1, 320, 256);
  WCONV(v2w,  wfv2, 256, 128);
  WCONV(q1w,  wfq1, 128, 256);
  WCONV(q2w,  wfq2, 256, 128);
  WCONV(o1w,  wfo1, 128, 256);
  WCONV(o2w,  wfo2, 256, 128);
  WCONV(ntw,  wfnt, 128, 256);
  WCONV(etw,  wfet, 128, 128);
  WCONV(f1w,  wff1, 128, 256);
  WCONV(f2w,  wff2, 256, 128);
#undef WCONV

  k_init<<<(NN*ND + 255)/256, 256, 0, stream>>>(outnum, segsum, segmax);

  const int nb_nodes = (NN + 15) / 16;
  k_node_a<<<nb_nodes, 128, 0, stream>>>(h, ntime, ntb, q1b, qng, qnb, q2b,
                                         wfnt, wfq1, wfq2, hmbf, nodesums, qbuf);

  k_edge_b<<<NE/16, 128, 0, stream>>>(eidx, pos, clash, eattr, etime,
                                      etb, ee_b, ln_g, ln_b,
                                      k1b, kng, knb, k2b,
                                      v1b, vng, vnb, v2b,
                                      wfet, wfee, wfk1, wfk2, wfv1, wfv2,
                                      hmbf, nodesums, qbuf,
                                      vbuf, logits, segmax);

  k_edge_c<<<(int)(((long long)NE*HEADS + 255)/256), 256, 0, stream>>>(
      eidx, logits, segmax, vbuf, segsum, outnum);

  k_node_d<<<nb_nodes, 128, 0, stream>>>(h, outnum, segsum,
                                         o1b, o2b, f1b, f2b,
                                         wfo1, wfo2, wff1, wff2, out);

  k_pos<<<(NN*3 + 255)/256, 256, 0, stream>>>(pos, out);
}
